// AdaTTWSharedExpsUnit_60808146976785
// MI455X (gfx1250) — matrix operand order repaired and checked by reading, not compiled, not run on hardware
//
#include <hip/hip_runtime.h>
#include <hip/hip_bf16.h>

// ---------------------------------------------------------------------------
// Problem constants (from the reference):  B=8192, T=8, E=17, D=512
// IDX[e] = e >> 1   (e in [0,16])
// ---------------------------------------------------------------------------
#define BB 8192
#define TT 8
#define EE 17
#define DD 512

typedef __attribute__((ext_vector_type(16))) __bf16 v16bf;
typedef __attribute__((ext_vector_type(8)))  float  v8f;

union BFrag {
    v16bf v;
    uint4 q[2];           // two 128-bit halves -> two ds_load_b128 per fragment
};

__device__ __forceinline__ unsigned short f2bf(float f) {
    unsigned int u = __float_as_uint(f);
    u = (u + 0x7FFFu + ((u >> 16) & 1u)) >> 16;   // round-to-nearest-even
    return (unsigned short)u;
}
__device__ __forceinline__ float bf2f(unsigned short h) {
    return __uint_as_float(((unsigned int)h) << 16);
}

// ---------------------------------------------------------------------------
// fp32 -> bf16 conversion (one thread per element), row-major preserved (activations)
// ---------------------------------------------------------------------------
__global__ void cvt_bf16_kernel(const float* __restrict__ src,
                                unsigned short* __restrict__ dst, int n) {
    int i = blockIdx.x * blockDim.x + threadIdx.x;
    if (i < n) dst[i] = f2bf(src[i]);
}

// ---------------------------------------------------------------------------
// fp32 -> bf16 conversion + transpose:  src (E, K, N) -> dst (E, N, K)
// Makes the in-kernel weight panel staging a straight contiguous copy.
// ---------------------------------------------------------------------------
__global__ void cvt_t_bf16_kernel(const float* __restrict__ src,
                                  unsigned short* __restrict__ dst,
                                  int K, int N, long long total) {
    long long t = (long long)blockIdx.x * blockDim.x + threadIdx.x;
    if (t >= total) return;
    const int n = (int)(t % N);
    long long r = t / N;
    const int k = (int)(r % K);
    const int e = (int)(r / K);
    dst[((size_t)e * N + n) * K + k] = f2bf(src[t]);
}

// ---------------------------------------------------------------------------
// WMMA fragment loaders (16-bit data, per CDNA5 ISA 7.12.2 layouts).
// A (16x32, MxK), row-major in LDS:
//   lane: m = lane&15, half = lane>>4
//   VGPR j<4  : K = 2j,2j+1  (+ half*8)   -> row[half*8 + 0..7]    (one b128)
//   VGPR j>=4 : K = 16+2(j-4) (+ half*8)  -> row[16 + half*8 + 0..7] (one b128)
// B (32x16, KxN), stored transposed [n][k] in LDS:
//   lane: n = lane&15, half = lane>>4
//   same K order as A: row[half*8 + 0..7], row[16 + half*8 + 0..7] (two b128)
// All rows are 16B-aligned (ld is a multiple of 8 elements, k0 multiple of 32).
// ---------------------------------------------------------------------------
__device__ __forceinline__ v16bf load_a_frag(const unsigned short* sA, int ld,
                                             int k0, int lane) {
    const int half = lane >> 4;
    const int m    = lane & 15;
    const unsigned short* row = sA + m * ld + k0 + half * 8;
    BFrag f;
    f.q[0] = *(const uint4*)(row);
    f.q[1] = *(const uint4*)(row + 16);
    return f.v;
}

__device__ __forceinline__ v16bf load_b_frag(const unsigned short* sBt, int ld,
                                             int k0, int lane) {
    const int half = lane >> 4;
    const int n    = lane & 15;
    const unsigned short* row = sBt + n * ld + k0 + half * 8;
    BFrag f;
    f.q[0] = *(const uint4*)(row);
    f.q[1] = *(const uint4*)(row + 16);
    return f.v;
}

// ---------------------------------------------------------------------------
// Fused per-expert MLP:  E = relu( relu(X * W0 + b0) * W1 + b1 )
//   grid  = (B/64, 17), block = 256 threads = 8 waves
//   Waves arranged 2(M) x 4(N); each wave owns a 32x32 register tile
//   (2x2 of 16x16 accumulators) -> 4 WMMA per K-step off 4 fragment loads.
//   Output-column panels are 128 wide; weights are pre-transposed [N][K].
//   Dynamic LDS: sX[64*ldp] | sH[64*ldp] | sW[128*ldp]   (ushort elements)
// ---------------------------------------------------------------------------
__global__ void expert_ffn_kernel(const unsigned short* __restrict__ X,
                                  int xRowStride, int useIdx,
                                  const unsigned short* __restrict__ W0T,
                                  const float* __restrict__ B0,
                                  const unsigned short* __restrict__ W1T,
                                  const float* __restrict__ B1,
                                  unsigned short* __restrict__ Eout,
                                  int Din, int Dh, int Dout, int ldp) {
    extern __shared__ unsigned short smem[];
    unsigned short* sX = smem;                  // 64 x ldp  (input tile)
    unsigned short* sH = smem + 64 * ldp;       // 64 x ldp  (hidden tile)
    unsigned short* sW = smem + 128 * ldp;      // 128 x ldp (transposed W panel)

    const int tid  = threadIdx.x;
    const int lane = tid & 31;
    const int wave = tid >> 5;
    const int wm   = wave >> 2;   // 0..1 -> 32-row M block
    const int wn   = wave & 3;    // 0..3 -> 32-col N block within 128 panel

    const int e  = blockIdx.y;
    const int b0 = blockIdx.x * 64;
    const int xColOff = useIdx ? (e >> 1) * Din : 0;

    const unsigned short* W0Te = W0T + (size_t)e * Din * Dh;   // (Dh, Din)
    const unsigned short* W1Te = W1T + (size_t)e * Dh * Dout;  // (Dout, Dh)
    const float*          B0e  = B0 + (size_t)e * Dh;
    const float*          B1e  = B1 + (size_t)e * Dout;

    // ---- load X tile (64 x Din) into LDS with 128-bit copies -------------
    {
        const int r0   = tid >> 5;             // 0..7
        const int DinQ = Din >> 3;             // uint4 (8 bf16) per row
        for (int r = r0; r < 64; r += 8) {
            const uint4* src =
                (const uint4*)(X + (size_t)(b0 + r) * xRowStride + xColOff);
            uint4* dst = (uint4*)(sX + r * ldp);
            for (int c = lane; c < DinQ; c += 32) dst[c] = src[c];
        }
    }

    // ---- stage 1: H = relu(X * W0 + b0) -> LDS (bf16) --------------------
    const int nsteps1 = Dh >> 7;               // 128-wide panels
    for (int ns = 0; ns < nsteps1; ++ns) {
        const int n0p = ns << 7;
        __syncthreads();                       // sW free / sX ready (iter 0)
        {   // stage 128 rows of W0T (each row = Din contiguous bf16)
            const int DinQ = Din >> 3;
            for (int n = tid >> 5; n < 128; n += 8) {
                const uint4* s = (const uint4*)(W0Te + (size_t)(n0p + n) * Din);
                uint4* d = (uint4*)(sW + n * ldp);
                for (int c = lane; c < DinQ; c += 32) d[c] = s[c];
            }
        }
        __syncthreads();

        v8f acc00 = {}, acc01 = {}, acc10 = {}, acc11 = {};
        const unsigned short* aBase = sX + (32 * wm) * ldp;
        const unsigned short* bBase = sW + (32 * wn) * ldp;
        for (int k0 = 0; k0 < Din; k0 += 32) {
            v16bf a0 = load_a_frag(aBase,            ldp, k0, lane);
            v16bf a1 = load_a_frag(aBase + 16 * ldp, ldp, k0, lane);
            v16bf b0 = load_b_frag(bBase,            ldp, k0, lane);
            v16bf b1 = load_b_frag(bBase + 16 * ldp, ldp, k0, lane);
            acc00 = __builtin_amdgcn_wmma_f32_16x16x32_bf16(false, a0, false, b0, (short)0, acc00, false, false);
            acc01 = __builtin_amdgcn_wmma_f32_16x16x32_bf16(false, a0, false, b1, (short)0, acc01, false, false);
            acc10 = __builtin_amdgcn_wmma_f32_16x16x32_bf16(false, a1, false, b0, (short)0, acc10, false, false);
            acc11 = __builtin_amdgcn_wmma_f32_16x16x32_bf16(false, a1, false, b1, (short)0, acc11, false, false);
        }

        const int half = lane >> 4;
        const int nl   = lane & 15;
#pragma unroll
        for (int tn = 0; tn < 2; ++tn) {
            const int n_g  = n0p + 32 * wn + 16 * tn + nl;
            const float bias = B0e[n_g];
#pragma unroll
            for (int tm = 0; tm < 2; ++tm) {
                const v8f& a = tm ? (tn ? acc11 : acc10) : (tn ? acc01 : acc00);
#pragma unroll
                for (int r = 0; r < 8; ++r) {
                    const int m = 32 * wm + 16 * tm + r + 8 * half;
                    float v = a[r] + bias;
                    v = v > 0.0f ? v : 0.0f;
                    sH[m * ldp + n_g] = f2bf(v);
                }
            }
        }
    }
    __syncthreads();                            // hidden tile complete

    // ---- stage 2: E = relu(H * W1 + b1) -> global ------------------------
    const int nsteps2 = Dout >> 7;
    for (int ns = 0; ns < nsteps2; ++ns) {
        const int n0p = ns << 7;
        __syncthreads();
        {
            const int DhQ = Dh >> 3;
            for (int n = tid >> 5; n < 128; n += 8) {
                const uint4* s = (const uint4*)(W1Te + (size_t)(n0p + n) * Dh);
                uint4* d = (uint4*)(sW + n * ldp);
                for (int c = lane; c < DhQ; c += 32) d[c] = s[c];
            }
        }
        __syncthreads();

        v8f acc00 = {}, acc01 = {}, acc10 = {}, acc11 = {};
        const unsigned short* aBase = sH + (32 * wm) * ldp;
        const unsigned short* bBase = sW + (32 * wn) * ldp;
        for (int k0 = 0; k0 < Dh; k0 += 32) {
            v16bf a0 = load_a_frag(aBase,            ldp, k0, lane);
            v16bf a1 = load_a_frag(aBase + 16 * ldp, ldp, k0, lane);
            v16bf b0 = load_b_frag(bBase,            ldp, k0, lane);
            v16bf b1 = load_b_frag(bBase + 16 * ldp, ldp, k0, lane);
            acc00 = __builtin_amdgcn_wmma_f32_16x16x32_bf16(false, a0, false, b0, (short)0, acc00, false, false);
            acc01 = __builtin_amdgcn_wmma_f32_16x16x32_bf16(false, a0, false, b1, (short)0, acc01, false, false);
            acc10 = __builtin_amdgcn_wmma_f32_16x16x32_bf16(false, a1, false, b0, (short)0, acc10, false, false);
            acc11 = __builtin_amdgcn_wmma_f32_16x16x32_bf16(false, a1, false, b1, (short)0, acc11, false, false);
        }

        const int half = lane >> 4;
        const int nl   = lane & 15;
#pragma unroll
        for (int tn = 0; tn < 2; ++tn) {
            const int n_g  = n0p + 32 * wn + 16 * tn + nl;
            const float bias = B1e[n_g];
#pragma unroll
            for (int tm = 0; tm < 2; ++tm) {
                const v8f& a = tm ? (tn ? acc11 : acc10) : (tn ? acc01 : acc00);
#pragma unroll
                for (int r = 0; r < 8; ++r) {
                    const int m = 32 * wm + 16 * tm + r + 8 * half;
                    float v = a[r] + bias;
                    v = v > 0.0f ? v : 0.0f;
                    Eout[((size_t)(b0 + m) * EE + e) * Dout + n_g] = f2bf(v);
                }
            }
        }
    }
}

// ---------------------------------------------------------------------------
// gate0: g0[b,m,:] = softmax(x[b,:] . g0_w[m,:,:] + g0_b[m,:]) + res0[m,:]
//   one thread per (b, m);  m in [0, 9)
// ---------------------------------------------------------------------------
__global__ void gate0_kernel(const float* __restrict__ x,
                             const float* __restrict__ gw,
                             const float* __restrict__ gb,
                             const float* __restrict__ sew_task,
                             const float* __restrict__ sew_shared,
                             float* __restrict__ g0) {
    int t = blockIdx.x * blockDim.x + threadIdx.x;
    if (t >= BB * (TT + 1)) return;
    const int b = t / (TT + 1);
    const int m = t % (TT + 1);

    float acc[EE];
#pragma unroll
    for (int e = 0; e < EE; ++e) acc[e] = gb[m * EE + e];

    const float* xr = x + (size_t)b * DD;
    const float* wbase = gw + (size_t)m * DD * EE;
    for (int d = 0; d < DD; ++d) {
        const float xv = xr[d];
        const float* wr = wbase + (size_t)d * EE;
#pragma unroll
        for (int e = 0; e < EE; ++e) acc[e] += xv * wr[e];
    }
    float mx = acc[0];
#pragma unroll
    for (int e = 1; e < EE; ++e) mx = fmaxf(mx, acc[e]);
    float s = 0.0f;
#pragma unroll
    for (int e = 0; e < EE; ++e) { acc[e] = __expf(acc[e] - mx); s += acc[e]; }
    const float inv = 1.0f / s;
#pragma unroll
    for (int e = 0; e < EE; ++e) acc[e] *= inv;
    if (m < TT) {
        acc[2 * m]     += sew_task[m * 4 + 0];
        acc[2 * m + 1] += sew_task[m * 4 + 1];
    } else {
        acc[2 * TT]    += sew_shared[0];
    }
    float* out = g0 + (size_t)t * EE;
#pragma unroll
    for (int e = 0; e < EE; ++e) out[e] = acc[e];
}

// ---------------------------------------------------------------------------
// gate1: g1[b,m,:] = softmax(h1[b,m,:] . g1_w[m,:,:] + g1_b[m,:]) + res1[m,:]
//   one thread per (b, m);  m in [0, 8);  h1 is bf16 (B,9,256)
// ---------------------------------------------------------------------------
__global__ void gate1_kernel(const unsigned short* __restrict__ h1,
                             const float* __restrict__ gw,
                             const float* __restrict__ gb,
                             const float* __restrict__ sew_task,
                             float* __restrict__ g1) {
    int t = blockIdx.x * blockDim.x + threadIdx.x;
    if (t >= BB * TT) return;
    const int b = t / TT;
    const int m = t % TT;

    float acc[EE];
#pragma unroll
    for (int e = 0; e < EE; ++e) acc[e] = gb[m * EE + e];

    const unsigned short* hr = h1 + ((size_t)b * (TT + 1) + m) * 256;
    const float* wbase = gw + (size_t)m * 256 * EE;
    for (int d = 0; d < 256; ++d) {
        const float hv = bf2f(hr[d]);
        const float* wr = wbase + (size_t)d * EE;
#pragma unroll
        for (int e = 0; e < EE; ++e) acc[e] += hv * wr[e];
    }
    float mx = acc[0];
#pragma unroll
    for (int e = 1; e < EE; ++e) mx = fmaxf(mx, acc[e]);
    float s = 0.0f;
#pragma unroll
    for (int e = 0; e < EE; ++e) { acc[e] = __expf(acc[e] - mx); s += acc[e]; }
    const float inv = 1.0f / s;
#pragma unroll
    for (int e = 0; e < EE; ++e) acc[e] *= inv;
    acc[2 * m]     += sew_task[m * 4 + 2];
    acc[2 * m + 1] += sew_task[m * 4 + 3];
    float* out = g1 + (size_t)t * EE;
#pragma unroll
    for (int e = 0; e < EE; ++e) out[e] = acc[e];
}

// ---------------------------------------------------------------------------
// mix0: h1[b,m,k] = sum_e g0[b,m,e] * e0[b,e,k]   (m < 9) -> bf16
// ---------------------------------------------------------------------------
__global__ void mix0_kernel(const float* __restrict__ g,
                            const unsigned short* __restrict__ ebuf,
                            unsigned short* __restrict__ hout) {
    const long long t = (long long)blockIdx.x * blockDim.x + threadIdx.x;
    if (t >= (long long)BB * (TT + 1) * 256) return;
    const int k = (int)(t & 255);
    const long long bm = t >> 8;
    const int m = (int)(bm % (TT + 1));
    const long long b = bm / (TT + 1);

    const float* gr = g + (b * (TT + 1) + m) * EE;
    const unsigned short* er = ebuf + (b * EE) * 256 + k;
    float s = 0.0f;
#pragma unroll
    for (int e = 0; e < EE; ++e) s += gr[e] * bf2f(er[(size_t)e * 256]);
    hout[t] = f2bf(s);
}

// ---------------------------------------------------------------------------
// mix1: out[b,m,k] = sum_e g1[b,m,e] * e1[b,e,k]   (m < 8) -> fp32 (d_out)
// ---------------------------------------------------------------------------
__global__ void mix1_kernel(const float* __restrict__ g,
                            const unsigned short* __restrict__ ebuf,
                            float* __restrict__ out) {
    const long long t = (long long)blockIdx.x * blockDim.x + threadIdx.x;
    if (t >= (long long)BB * TT * 256) return;
    const int k = (int)(t & 255);
    const long long bm = t >> 8;
    const int m = (int)(bm % TT);
    const long long b = bm / TT;

    const float* gr = g + (b * TT + m) * EE;
    const unsigned short* er = ebuf + (b * EE) * 256 + k;
    float s = 0.0f;
#pragma unroll
    for (int e = 0; e < EE; ++e) s += gr[e] * bf2f(er[(size_t)e * 256]);
    out[t] = s;
}

// ---------------------------------------------------------------------------
// Host orchestration
// ---------------------------------------------------------------------------
extern "C" void kernel_launch(void* const* d_in, const int* in_sizes, int n_in,
                              void* d_out, int out_size, void* d_ws, size_t ws_size,
                              hipStream_t stream) {
    const float* x    = (const float*)d_in[0];
    const float* l0w0 = (const float*)d_in[1];
    const float* l0b0 = (const float*)d_in[2];
    const float* l0w1 = (const float*)d_in[3];
    const float* l0b1 = (const float*)d_in[4];
    const float* l1w0 = (const float*)d_in[5];
    const float* l1b0 = (const float*)d_in[6];
    const float* l1w1 = (const float*)d_in[7];
    const float* l1b1 = (const float*)d_in[8];
    const float* g0w  = (const float*)d_in[9];
    const float* g0b  = (const float*)d_in[10];
    const float* g1w  = (const float*)d_in[11];
    const float* g1b  = (const float*)d_in[12];
    const float* sewt = (const float*)d_in[13];
    const float* sews = (const float*)d_in[14];

    char* ws = (char*)d_ws;
    size_t off = 0;
    auto carve = [&](size_t bytes) -> char* {
        char* p = ws + off;
        off = (off + bytes + 255) & ~(size_t)255;
        return p;
    };

    const size_t nX    = (size_t)BB * DD;
    const size_t nW0L0 = (size_t)EE * 512 * 512;
    const size_t nW1L0 = (size_t)EE * 512 * 256;
    const size_t nW0L1 = (size_t)EE * 256 * 256;
    const size_t nW1L1 = (size_t)EE * 256 * 256;

    unsigned short* x_bf   = (unsigned short*)carve(nX * 2);
    unsigned short* w0l0T  = (unsigned short*)carve(nW0L0 * 2);  // (E, Dh, Din)
    unsigned short* w1l0T  = (unsigned short*)carve(nW1L0 * 2);  // (E, Dout, Dh)
    unsigned short* w0l1T  = (unsigned short*)carve(nW0L1 * 2);
    unsigned short* w1l1T  = (unsigned short*)carve(nW1L1 * 2);
    unsigned short* e_buf  = (unsigned short*)carve((size_t)BB * EE * 256 * 2);
    unsigned short* h1_bf  = (unsigned short*)carve((size_t)BB * (TT + 1) * 256 * 2);
    float*          g0_buf = (float*)carve((size_t)BB * (TT + 1) * EE * 4);
    float*          g1_buf = (float*)carve((size_t)BB * TT * EE * 4);
    (void)ws_size; (void)in_sizes; (void)n_in; (void)out_size;

    // 1) convert activations (row-major) + weights (transposed) to bf16
    cvt_bf16_kernel<<<(unsigned)((nX + 255) / 256), 256, 0, stream>>>(
        x, x_bf, (int)nX);
    auto cvtT = [&](const float* s, unsigned short* d, int K, int N) {
        const long long n = (long long)EE * K * N;
        cvt_t_bf16_kernel<<<(unsigned)((n + 255) / 256), 256, 0, stream>>>(
            s, d, K, N, n);
    };
    cvtT(l0w0, w0l0T, 512, 512);
    cvtT(l0w1, w1l0T, 512, 256);
    cvtT(l1w0, w0l1T, 256, 256);
    cvtT(l1w1, w1l1T, 256, 256);

    // 2) gate 0
    gate0_kernel<<<(BB * (TT + 1) + 255) / 256, 256, 0, stream>>>(
        x, g0w, g0b, sewt, sews, g0_buf);

    // 3) layer-0 experts (512 -> 512 -> 256)
    const int ldp0 = 520;                               // pad, multiple of 8
    const size_t smem0 = (size_t)256 * ldp0 * 2;        // 266,240 B
    (void)hipFuncSetAttribute((const void*)expert_ffn_kernel,
                              hipFuncAttributeMaxDynamicSharedMemorySize,
                              (int)smem0);
    expert_ffn_kernel<<<dim3(BB / 64, EE), 256, smem0, stream>>>(
        x_bf, DD, /*useIdx=*/0, w0l0T, l0b0, w1l0T, l0b1, e_buf,
        /*Din=*/512, /*Dh=*/512, /*Dout=*/256, ldp0);

    // 4) mix 0 -> h1 (B, 9, 256) bf16
    {
        const long long n = (long long)BB * (TT + 1) * 256;
        mix0_kernel<<<(unsigned)((n + 255) / 256), 256, 0, stream>>>(
            g0_buf, e_buf, h1_bf);
    }

    // 5) gate 1
    gate1_kernel<<<(BB * TT + 255) / 256, 256, 0, stream>>>(
        h1_bf, g1w, g1b, sewt, g1_buf);

    // 6) layer-1 experts (256 -> 256 -> 256), rows gathered via IDX[e] = e>>1
    const int ldp1 = 264;
    const size_t smem1 = (size_t)256 * ldp1 * 2;        // 135,168 B
    expert_ffn_kernel<<<dim3(BB / 64, EE), 256, smem1, stream>>>(
        h1_bf, (TT + 1) * 256, /*useIdx=*/1, w0l1T, l1b0, w1l1T, l1b1, e_buf,
        /*Din=*/256, /*Dh=*/256, /*Dout=*/256, ldp1);

    // 7) mix 1 -> d_out (B, 8, 256) fp32
    {
        const long long n = (long long)BB * TT * 256;
        mix1_kernel<<<(unsigned)((n + 255) / 256), 256, 0, stream>>>(
            g1_buf, e_buf, (float*)d_out);
    }
}
